// MultiHeadedAttention_30812095381859
// MI455X (gfx1250) — compile-verified, matrix operand order checked by reading, not run on hardware
//
#include <hip/hip_runtime.h>
#include <hip/hip_bf16.h>
#include <stdint.h>

// ---------------------------------------------------------------------------
// MultiHeadedAttention for MI455X (gfx1250): f16 WMMA everywhere, fp32 accum.
// D_MODEL=1024, H=16, D_K=64, B=2, S=2048.
// ---------------------------------------------------------------------------

typedef _Float16 h16v __attribute__((ext_vector_type(16)));
typedef _Float16 h8v  __attribute__((ext_vector_type(8)));
typedef _Float16 h4v  __attribute__((ext_vector_type(4)));
typedef float    f8v  __attribute__((ext_vector_type(8)));
typedef float    f4v  __attribute__((ext_vector_type(4)));

#define D_MODEL 1024
#define SEQ     2048
#define NHEAD   16
#define DK      64
#define BSZ     2

// Scheduling pipeline hints: force [DS-read][VMEM-read][WMMA] grouping so all
// B-fragment loads stay in flight instead of being serialized onto one
// register set (mask: 0x8=MFMA/WMMA, 0x20=VMEM read, 0x100=DS read, 0x200=DS write).
#if __has_builtin(__builtin_amdgcn_sched_group_barrier)
#define SGB(mask, cnt) __builtin_amdgcn_sched_group_barrier((mask), (cnt), 0)
#else
#define SGB(mask, cnt)
#endif

// WMMA f32 += f16 x f16, 16x16x32
__device__ inline f8v wmma_f16(h16v a, h16v b, f8v c) {
  return __builtin_amdgcn_wmma_f32_16x16x32_f16(
      /*neg_a=*/false, a, /*neg_b=*/false, b,
      /*c_mod=*/(short)0, c, /*reuse_a=*/false, /*reuse_b=*/false);
}

// A/B fragment loader: per-lane two contiguous 8-half chunks at +0 and +16
// (CDNA5 16-bit A 16x32 layout: lanes<16 hold K=kb..kb+7 & kb+16..kb+23).
// p must already point at (row, k_base + kb); 16B aligned.
__device__ inline h16v frag_ld_h(const _Float16* p) {
  h8v lo = *(const h8v*)p;
  h8v hi = *(const h8v*)(p + 16);
  h16v r;
#pragma unroll
  for (int i = 0; i < 8; ++i) { r[i] = lo[i]; r[i + 8] = hi[i]; }
  return r;
}

// Same fragment shape, but source is fp32 (LDS softmax probs) -> cvt to f16.
__device__ inline h16v frag_ld_f32(const float* p) {
  h16v r;
#pragma unroll
  for (int i = 0; i < 8; ++i) {
    r[i]     = (_Float16)p[i];
    r[i + 8] = (_Float16)p[i + 16];
  }
  return r;
}

// Async-copy a contiguous `bytes` block global -> LDS, whole block (256 thr).
// Per lane: 16B chunks via global_load_async_to_lds_b128 (ASYNCcnt).
__device__ inline void async_stage(const _Float16* gsrc, _Float16* lds_dst,
                                   int bytes, int tid) {
  unsigned lbase = (unsigned)(uintptr_t)lds_dst;            // LDS offset
  unsigned long long gbase = (unsigned long long)(uintptr_t)gsrc;
  int chunks = bytes >> 4;                                  // 16B chunks
  for (int c = tid; c < chunks; c += 256) {
    unsigned ldsa = lbase + ((unsigned)c << 4);
    unsigned long long ga = gbase + ((unsigned long long)c << 4);
    asm volatile("global_load_async_to_lds_b128 %0, %1, off"
                 :: "v"(ldsa), "v"(ga) : "memory");
  }
  asm volatile("s_wait_asynccnt 0x0" ::: "memory");
}

// ---------------------------------------------------------------------------
// fp32 -> f16 conversion pass (vectorized: 4 elems/thread)
// ---------------------------------------------------------------------------
__global__ void cvt_f32_f16(const f4v* __restrict__ in,
                            h4v* __restrict__ out, int n4) {
  int i = blockIdx.x * blockDim.x + threadIdx.x;
  if (i < n4) {
    f4v v = in[i];
    h4v o;
#pragma unroll
    for (int j = 0; j < 4; ++j) o[j] = (_Float16)v[j];
    out[i] = o;
  }
}

// ---------------------------------------------------------------------------
// Projection GEMM: out[m,n] = sum_k X[m,k]*W[n,k] + bias[n]
// M = 4096 (16 rows/block), N = 1024 (8 waves x 8 n-tiles), K = 1024.
// A tile (16x1024 f16 = 32KB, contiguous) async-staged in LDS once per block.
// vt_mode==1: scatter-store V transposed per head: Vt[b,h,d,s] (f16).
// ---------------------------------------------------------------------------
__global__ __launch_bounds__(256) void proj_gemm(
    const _Float16* __restrict__ X, const _Float16* __restrict__ W,
    const float* __restrict__ bias, _Float16* __restrict__ out, int vt_mode) {
  __shared__ _Float16 sA[16 * D_MODEL];  // 32KB

  const int tid = threadIdx.x;
  const int w = tid >> 5, lane = tid & 31, lrow = lane & 15;
  const int kb = (lane < 16) ? 0 : 8;
  const int mh = (lane < 16) ? 0 : 8;
  const int r0 = blockIdx.x * 16;

  async_stage(X + (size_t)r0 * D_MODEL, sA, 16 * D_MODEL * 2, tid);
  __syncthreads();

  f8v acc[8] = {};
  const _Float16* arow = sA + lrow * D_MODEL + kb;

  for (int ks = 0; ks < 32; ++ks) {
    h16v a = frag_ld_h(arow + ks * 32);          // LDS: 2 ds_load_b128
    h16v bfr[8];
#pragma unroll
    for (int i = 0; i < 8; ++i) {                // 16 global_load_b128 in flight
      int n = (w * 8 + i) * 16 + lrow;
      bfr[i] = frag_ld_h(W + (size_t)n * D_MODEL + ks * 32 + kb);
    }
#pragma unroll
    for (int i = 0; i < 8; ++i)
      acc[i] = wmma_f16(a, bfr[i], acc[i]);
    SGB(0x100, 2);   // DS reads (A fragment)
    SGB(0x020, 16);  // VMEM reads (all 8 B fragments)
    SGB(0x008, 8);   // 8 WMMAs
  }

#pragma unroll
  for (int i = 0; i < 8; ++i) {
    int nt = w * 8 + i;
#pragma unroll
    for (int r = 0; r < 8; ++r) {
      int m = r0 + r + mh;           // global row (b*S + s)
      int n = nt * 16 + lrow;        // output feature
      float v = acc[i][r] + bias[n];
      if (!vt_mode) {
        out[(size_t)m * D_MODEL + n] = (_Float16)v;
      } else {
        int b = m >> 11, s = m & (SEQ - 1);
        int h = n >> 6, dd = n & (DK - 1);
        out[((size_t)(b * NHEAD + h) * DK + dd) * SEQ + s] = (_Float16)v;
      }
    }
  }
}

// ---------------------------------------------------------------------------
// Fused attention: scores (WMMA) -> softmax in LDS -> attn out (HBM, NT) ->
// context = P x Vt (WMMA) -> Ctx f16.
// One workgroup per (b, h, 16-query tile). LDS: 16x2048 f32 scores + 16x64 ctx.
// ---------------------------------------------------------------------------
__global__ __launch_bounds__(256) void attn_kernel(
    const _Float16* __restrict__ Qh, const _Float16* __restrict__ Kh,
    const _Float16* __restrict__ Vt, _Float16* __restrict__ Ctx,
    float* __restrict__ attn_out) {
  extern __shared__ char dyn_smem[];
  float* sm  = (float*)dyn_smem;                       // [16][2048] scores/probs
  float* ctx = (float*)(dyn_smem + 16 * SEQ * 4);      // [16][64] context accum

  const int tid = threadIdx.x;
  const int w = tid >> 5, lane = tid & 31, lrow = lane & 15;
  const int kb = (lane < 16) ? 0 : 8;
  const int mh = (lane < 16) ? 0 : 8;

  const int idx = blockIdx.x;
  const int qt = idx & 127;
  const int h  = (idx >> 7) & (NHEAD - 1);
  const int b  = idx >> 11;
  const int q0 = qt * 16;

  for (int i = tid; i < 16 * DK; i += 256) ctx[i] = 0.f;

  // ---- Phase 1: scores[q, k] = (Q . K) / sqrt(DK), into LDS ----
  const _Float16* qbase =
      Qh + (size_t)(b * SEQ + q0 + lrow) * D_MODEL + h * DK + kb;
  h16v a0 = frag_ld_h(qbase);        // d = 0..31
  h16v a1 = frag_ld_h(qbase + 32);   // d = 32..63

  for (int i = 0; i < 16; ++i) {     // 8 waves x 16 tiles = 128 key tiles
    int k0 = (w * 16 + i) * 16;
    const _Float16* kbase =
        Kh + (size_t)(b * SEQ + k0 + lrow) * D_MODEL + h * DK + kb;
    h16v b0 = frag_ld_h(kbase);
    h16v b1 = frag_ld_h(kbase + 32);
    f8v c = {};
    c = wmma_f16(a0, b0, c);
    c = wmma_f16(a1, b1, c);
#pragma unroll
    for (int r = 0; r < 8; ++r)
      sm[(r + mh) * SEQ + k0 + lrow] = c[r] * 0.125f;  // 1/sqrt(64)
    SGB(0x020, 4);   // VMEM reads (K fragments)
    SGB(0x008, 2);   // 2 WMMAs
    SGB(0x200, 8);   // 8 DS writes (scores)
  }
  __syncthreads();

  // ---- Phase 2: row softmax, write normalized probs to LDS + HBM (NT) ----
  float* arow_out = attn_out + ((size_t)((b * NHEAD + h) * SEQ + q0)) * SEQ;
  for (int rr = 0; rr < 2; ++rr) {
    int r = w * 2 + rr;
    float* row = sm + r * SEQ;
    float mx = -3.4e38f;
    for (int c = lane; c < SEQ; c += 32) mx = fmaxf(mx, row[c]);
    for (int o = 16; o > 0; o >>= 1) mx = fmaxf(mx, __shfl_xor(mx, o));
    float s = 0.f;
    for (int c = lane; c < SEQ; c += 32) {
      float e = __expf(row[c] - mx);
      row[c] = e;
      s += e;
    }
    for (int o = 16; o > 0; o >>= 1) s += __shfl_xor(s, o);
    float inv = 1.f / s;
    float* gout = arow_out + (size_t)r * SEQ;
    for (int c = lane; c < SEQ; c += 32) {
      float p = row[c] * inv;
      row[c] = p;
      __builtin_nontemporal_store(p, gout + c);  // 536MB stream: bypass L2 reuse
    }
  }
  __syncthreads();

  // ---- Phase 3: context = P[16x2048] x V[2048x64] via Vt ----
  const int d0 = (w & 3) * 16;   // which 16-wide d chunk
  const int kh = w >> 2;         // which half of the j (key) range
  size_t vtb = ((size_t)(b * NHEAD + h) * DK + d0 + lrow) * SEQ;
  f8v cacc = {};
  for (int ks = 0; ks < 32; ++ks) {
    int jb = (kh * 32 + ks) * 32 + kb;
    h16v a  = frag_ld_f32(sm + lrow * SEQ + jb);   // probs f32 -> f16
    h16v bf = frag_ld_h(Vt + vtb + jb);            // Vt row d, contiguous in j
    cacc = wmma_f16(a, bf, cacc);
  }
#pragma unroll
  for (int r = 0; r < 8; ++r)
    atomicAdd(&ctx[(r + mh) * DK + d0 + lrow], cacc[r]);  // ds_add_f32
  __syncthreads();

  for (int i = tid; i < 16 * DK; i += 256) {
    int m = i >> 6, dd = i & (DK - 1);
    Ctx[(size_t)(b * SEQ + q0 + m) * D_MODEL + h * DK + dd] = (_Float16)ctx[i];
  }
}

// ---------------------------------------------------------------------------
// Output projection + bias + residual + LayerNorm, fused.
// 16 rows/block, N=1024 (8 waves x 8 n-tiles), K=1024.
// Ctx A-tile async-staged in LDS (32KB static); x[16][1024] in dynamic LDS.
// ---------------------------------------------------------------------------
__global__ __launch_bounds__(256) void oproj_ln(
    const _Float16* __restrict__ Ctx, const _Float16* __restrict__ Wo,
    const float* __restrict__ bo, const float* __restrict__ resid,
    const float* __restrict__ gamma, const float* __restrict__ beta,
    float* __restrict__ outp) {
  __shared__ _Float16 sA[16 * D_MODEL];  // 32KB
  extern __shared__ char dyn_smem[];
  float* xs = (float*)dyn_smem;          // [16][1024]

  const int tid = threadIdx.x;
  const int w = tid >> 5, lane = tid & 31, lrow = lane & 15;
  const int kb = (lane < 16) ? 0 : 8;
  const int mh = (lane < 16) ? 0 : 8;
  const int r0 = blockIdx.x * 16;

  async_stage(Ctx + (size_t)r0 * D_MODEL, sA, 16 * D_MODEL * 2, tid);
  __syncthreads();

  f8v acc[8] = {};
  const _Float16* arow = sA + lrow * D_MODEL + kb;
  for (int ks = 0; ks < 32; ++ks) {
    h16v a = frag_ld_h(arow + ks * 32);
    h16v bfr[8];
#pragma unroll
    for (int i = 0; i < 8; ++i) {
      int n = (w * 8 + i) * 16 + lrow;
      bfr[i] = frag_ld_h(Wo + (size_t)n * D_MODEL + ks * 32 + kb);
    }
#pragma unroll
    for (int i = 0; i < 8; ++i)
      acc[i] = wmma_f16(a, bfr[i], acc[i]);
    SGB(0x100, 2);   // DS reads (A fragment)
    SGB(0x020, 16);  // VMEM reads (all 8 B fragments)
    SGB(0x008, 8);   // 8 WMMAs
  }

#pragma unroll
  for (int i = 0; i < 8; ++i) {
    int n = (w * 8 + i) * 16 + lrow;
#pragma unroll
    for (int r = 0; r < 8; ++r) {
      int m = r + mh;
      xs[m * D_MODEL + n] =
          acc[i][r] + bo[n] + resid[(size_t)(r0 + m) * D_MODEL + n];
    }
  }
  __syncthreads();

  for (int rr = 0; rr < 2; ++rr) {
    int r = w * 2 + rr;
    const float* row = xs + r * D_MODEL;
    float s = 0.f, s2 = 0.f;
    for (int c = lane; c < D_MODEL; c += 32) {
      float v = row[c];
      s += v;
      s2 += v * v;
    }
    for (int o = 16; o > 0; o >>= 1) {
      s  += __shfl_xor(s, o);
      s2 += __shfl_xor(s2, o);
    }
    float mu = s * (1.f / D_MODEL);
    float var = s2 * (1.f / D_MODEL) - mu * mu;
    float rstd = rsqrtf(var + 1e-5f);
    float* gout = outp + (size_t)(r0 + r) * D_MODEL;
    for (int c = lane; c < D_MODEL; c += 32)
      gout[c] = (row[c] - mu) * rstd * gamma[c] + beta[c];
  }
}

// ---------------------------------------------------------------------------
// Host launcher
// ---------------------------------------------------------------------------
extern "C" void kernel_launch(void* const* d_in, const int* in_sizes, int n_in,
                              void* d_out, int out_size, void* d_ws,
                              size_t ws_size, hipStream_t stream) {
  const float* query = (const float*)d_in[0];
  const float* key_i = (const float*)d_in[1];
  const float* value = (const float*)d_in[2];
  // d_in[3] = attn_mask: faithfully ignored (reference discards masked_fill)
  const float* Wq = (const float*)d_in[4];
  const float* bq = (const float*)d_in[5];
  const float* Wk = (const float*)d_in[6];
  const float* bk = (const float*)d_in[7];
  const float* Wv = (const float*)d_in[8];
  const float* bv = (const float*)d_in[9];
  const float* Wo = (const float*)d_in[10];
  const float* bo = (const float*)d_in[11];
  const float* gamma = (const float*)d_in[12];
  const float* beta  = (const float*)d_in[13];

  const int NX = BSZ * SEQ * D_MODEL;   // 4194304
  const int NW = D_MODEL * D_MODEL;     // 1048576

  _Float16* Xq  = (_Float16*)d_ws;
  _Float16* Xk  = Xq + NX;
  _Float16* Xv  = Xk + NX;
  _Float16* Wqh = Xv + NX;
  _Float16* Wkh = Wqh + NW;
  _Float16* Wvh = Wkh + NW;
  _Float16* Woh = Wvh + NW;
  _Float16* Qh  = Woh + NW;
  _Float16* Kh  = Qh + NX;
  _Float16* Vt  = Kh + NX;              // [B,H,DK,S]
  _Float16* Ctx = Vt + NX;

  float* out_ln = (float*)d_out;
  float* attn   = out_ln + (size_t)NX;  // [B,H,S,S] fp32

  cvt_f32_f16<<<NX / 4 / 256, 256, 0, stream>>>((const f4v*)query, (h4v*)Xq, NX / 4);
  cvt_f32_f16<<<NX / 4 / 256, 256, 0, stream>>>((const f4v*)key_i, (h4v*)Xk, NX / 4);
  cvt_f32_f16<<<NX / 4 / 256, 256, 0, stream>>>((const f4v*)value, (h4v*)Xv, NX / 4);
  cvt_f32_f16<<<NW / 4 / 256, 256, 0, stream>>>((const f4v*)Wq, (h4v*)Wqh, NW / 4);
  cvt_f32_f16<<<NW / 4 / 256, 256, 0, stream>>>((const f4v*)Wk, (h4v*)Wkh, NW / 4);
  cvt_f32_f16<<<NW / 4 / 256, 256, 0, stream>>>((const f4v*)Wv, (h4v*)Wvh, NW / 4);
  cvt_f32_f16<<<NW / 4 / 256, 256, 0, stream>>>((const f4v*)Wo, (h4v*)Woh, NW / 4);

  proj_gemm<<<BSZ * SEQ / 16, 256, 0, stream>>>(Xq, Wqh, bq, Qh, 0);
  proj_gemm<<<BSZ * SEQ / 16, 256, 0, stream>>>(Xk, Wkh, bk, Kh, 0);
  proj_gemm<<<BSZ * SEQ / 16, 256, 0, stream>>>(Xv, Wvh, bv, Vt, 1);

  const size_t attn_lds = (size_t)16 * SEQ * 4 + (size_t)16 * DK * 4; // 132 KB
  attn_kernel<<<BSZ * NHEAD * (SEQ / 16), 256, attn_lds, stream>>>(
      Qh, Kh, Vt, Ctx, attn);

  const size_t ln_lds = (size_t)16 * D_MODEL * 4;  // 64 KB dynamic (+32KB static)
  oproj_ln<<<BSZ * SEQ / 16, 256, ln_lds, stream>>>(Ctx, Woh, bo, query, gamma,
                                                    beta, out_ln);
}
